// LSTM_72962904424715
// MI455X (gfx1250) — matrix operand order repaired and checked by reading, not compiled, not run on hardware
//
#include <hip/hip_runtime.h>

// Problem dims
#define B_ 256
#define T_ 168
#define H_ 512
#define I_ 256

typedef unsigned short u16;
typedef __attribute__((ext_vector_type(16))) __bf16 v16bf;
typedef __attribute__((ext_vector_type(8)))  float  v8f;

__device__ __forceinline__ u16 f2bf(float f) {
    unsigned u = __float_as_uint(f);
    return (u16)((u + 0x7FFFu + ((u >> 16) & 1u)) >> 16);  // round-to-nearest-even
}
__device__ __forceinline__ float bf2f(u16 h) {
    return __uint_as_float(((unsigned)h) << 16);
}
__device__ __forceinline__ float sigm(float x) {
    return 1.0f / (1.0f + __expf(-x));
}

// A-tile: row-major [M,K] bf16 activations. Per ISA 16-bit A 16x32 layout:
// lane L: row M = L%16 ; K-base = (L/16)*8 ; VGPR0..3 = K..K+7, VGPR4..7 = K+16..K+23.
__device__ __forceinline__ v16bf load_a_tile(const u16* __restrict__ base, int stride,
                                             int row0, int k0, int lane) {
    const u16* p = base + (long)(row0 + (lane & 15)) * stride + (k0 + ((lane >> 4) << 3));
    union { uint4 u[2]; v16bf v; } r;
    r.u[0] = *(const uint4*)(p);
    r.u[1] = *(const uint4*)(p + 16);
    return r.v;
}

// A-tile from a precomputed per-wave row pointer (already includes lane row + k phase).
__device__ __forceinline__ v16bf load_a_row(const u16* arow, int k0) {
    union { uint4 u[2]; v16bf v; } r;
    r.u[0] = *(const uint4*)(arow + k0);
    r.u[1] = *(const uint4*)(arow + k0 + 16);
    return r.v;
}

// B-tile for x @ W^T with W stored [N,K] row-major, loaded from global. Per ISA B striping:
// lane L: col N = L%16 ; K-base = (L/16)*8 ; elems 0..7 = K..K+7, elems 8..15 = K+16..K+23.
__device__ __forceinline__ v16bf load_b_tile(const u16* __restrict__ W, int stride,
                                             int col0, int k0, int lane) {
    const u16* p = W + (long)(col0 + (lane & 15)) * stride + (k0 + ((lane >> 4) << 3));
    union { uint4 u[2]; v16bf v; } r;
    r.u[0] = *(const uint4*)(p);
    r.u[1] = *(const uint4*)(p + 16);
    return r.v;
}

// B-tile from an LDS-staged [4 gates][16 cols][32 k] bf16 buffer.
__device__ __forceinline__ v16bf lds_b_tile(const u16* buf, int gate, int lane) {
    const u16* p = buf + (gate * 16 + (lane & 15)) * 32 + ((lane >> 4) << 3);
    union { uint4 u[2]; v16bf v; } r;
    r.u[0] = *(const uint4*)(p);
    r.u[1] = *(const uint4*)(p + 16);
    return r.v;
}

#define WMMA_BF16(A, Bm, C) \
    __builtin_amdgcn_wmma_f32_16x16x32_bf16(false, (A), false, (Bm), (short)0, (C), false, false)

#define ASYNC_STAGE(lo, g) \
    asm volatile("global_load_async_to_lds_b128 %0, %1, off" :: "v"(lo), "v"(g) : "memory")
#define WAIT_ASYNC1() asm volatile("s_wait_asynccnt 0x1" ::: "memory")
#define WAIT_ASYNC0() asm volatile("s_wait_asynccnt 0x0" ::: "memory")

// ---------------------------------------------------------------------------
// f32 -> bf16 conversion (weights)
__global__ void cvt_bf16_kernel(const float* __restrict__ in, u16* __restrict__ out, int n) {
    int i = blockIdx.x * 256 + threadIdx.x;
    if (i < n) out[i] = f2bf(in[i]);
}

// zero a buffer of 32-bit words
__global__ void zero_kernel(unsigned int* __restrict__ p, int nwords) {
    int i = blockIdx.x * 256 + threadIdx.x;
    if (i < nwords) p[i] = 0u;
}

// Embedding gather + concat -> bf16 [B*T, 256]. One block per (b,t) row.
__global__ void embed_kernel(const int* __restrict__ x,
                             const float* __restrict__ ed, const float* __restrict__ em,
                             const float* __restrict__ eh, const float* __restrict__ emin,
                             u16* __restrict__ e) {
    int row   = blockIdx.x;          // b*T + t
    int col   = threadIdx.x;         // 0..255
    int which = col >> 6;
    int sub   = col & 63;
    int id    = x[row * 4 + which];
    const float* tab = (which == 0) ? ed : (which == 1) ? em : (which == 2) ? eh : emin;
    e[(long)row * I_ + col] = f2bf(tab[id * 64 + sub]);
}

// ---------------------------------------------------------------------------
// One LSTM timestep: gates = x_t @ W_ih^T + h_prev @ W_hh^T + b ; cell update.
// Grid: 64 blocks x 256 threads. All 8 waves of a block share one hidden tile h0
// (different batch tiles m0); the 4-gate weight B-tile is block-shared, staged in
// LDS by double-buffered global_load_async_to_lds_b128 with s_wait_asynccnt.
__global__ __launch_bounds__(256) void lstm_step_kernel(
        const u16* __restrict__ x, int x_stride, int D,
        const u16* __restrict__ h_prev,
        const u16* __restrict__ W_ih, const u16* __restrict__ W_hh,
        const float* __restrict__ b_ih, const float* __restrict__ b_hh,
        float* __restrict__ c, u16* __restrict__ h_next,
        u16* __restrict__ hs_out, int hs_stride) {
    __shared__ u16 bsm[2][4 * 16 * 32];      // 2 x 4KB double-buffered B tiles

    const int lane = threadIdx.x & 31;
    const int wave = threadIdx.x >> 5;
    const int m0   = ((((int)blockIdx.x & 1) << 3) + wave) << 4;  // 16 batch tiles
    const int h0   = ((int)blockIdx.x >> 1) << 4;                 // 32 hidden tiles

    // staging role: 256 threads copy 4 gates x 16 rows x 64B (4 x 16B chunks).
    // All row-base addressing hoisted out of the loops.
    const int tt    = threadIdx.x;
    const int chunk = tt & 3;
    const int brow  = (tt >> 2) & 15;
    const int bgate = tt >> 6;
    const int wrow  = bgate * H_ + h0 + brow;                 // weight row index
    const u16* gih  = W_ih + (long)wrow * D  + chunk * 8;     // k advances along row
    const u16* ghh  = W_hh + (long)wrow * H_ + chunk * 8;
    u16* lds_slot0  = &bsm[0][(bgate * 16 + brow) * 32 + chunk * 8];
    u16* lds_slot1  = &bsm[1][(bgate * 16 + brow) * 32 + chunk * 8];
    const unsigned lo0 = (unsigned)(uintptr_t)lds_slot0;      // flat->LDS: addr[31:0]
    const unsigned lo1 = (unsigned)(uintptr_t)lds_slot1;

    // per-wave A row pointers (lane row + k phase folded in)
    const u16* ax = x      + (long)(m0 + (lane & 15)) * x_stride + ((lane >> 4) << 3);
    const u16* ah = h_prev + (long)(m0 + (lane & 15)) * H_       + ((lane >> 4) << 3);

    v8f acc0 = {}, acc1 = {}, acc2 = {}, acc3 = {};
    int buf = 0;

    ASYNC_STAGE(lo0, gih);                        // prefetch k=0 of W_ih

    // ---- loop 1: input projection (K over D) ----
    for (int k0 = 0; k0 < D; k0 += 32) {
        __syncthreads();                          // other buffer's readers done
        const unsigned lon = buf ? lo0 : lo1;
        if (k0 + 32 < D) ASYNC_STAGE(lon, gih + k0 + 32);
        else             ASYNC_STAGE(lon, ghh);   // hand off to recurrent loop
        WAIT_ASYNC1();                            // current chunk landed
        __syncthreads();

        v16bf a = load_a_row(ax, k0);
        const u16* bb = &bsm[buf][0];
        v16bf b0 = lds_b_tile(bb, 0, lane);
        v16bf b1 = lds_b_tile(bb, 1, lane);
        v16bf b2 = lds_b_tile(bb, 2, lane);
        v16bf b3 = lds_b_tile(bb, 3, lane);
        acc0 = WMMA_BF16(a, b0, acc0);
        acc1 = WMMA_BF16(a, b1, acc1);
        acc2 = WMMA_BF16(a, b2, acc2);
        acc3 = WMMA_BF16(a, b3, acc3);
        buf ^= 1;
    }

    // ---- loop 2: recurrent projection (K over H, constant 16 iters) ----
    for (int k0 = 0; k0 < H_; k0 += 32) {
        __syncthreads();
        if (k0 + 32 < H_) {
            const unsigned lon = buf ? lo0 : lo1;
            ASYNC_STAGE(lon, ghh + k0 + 32);
            WAIT_ASYNC1();
        } else {
            WAIT_ASYNC0();
        }
        __syncthreads();

        v16bf a = load_a_row(ah, k0);
        const u16* bb = &bsm[buf][0];
        v16bf b0 = lds_b_tile(bb, 0, lane);
        v16bf b1 = lds_b_tile(bb, 1, lane);
        v16bf b2 = lds_b_tile(bb, 2, lane);
        v16bf b3 = lds_b_tile(bb, 3, lane);
        acc0 = WMMA_BF16(a, b0, acc0);
        acc1 = WMMA_BF16(a, b1, acc1);
        acc2 = WMMA_BF16(a, b2, acc2);
        acc3 = WMMA_BF16(a, b3, acc3);
        buf ^= 1;
    }

    // epilogue: gate nonlinearity + cell update (f32), emit bf16 h
    int hid = h0 + (lane & 15);
    float bi0 = b_ih[hid]        + b_hh[hid];
    float bi1 = b_ih[512 + hid]  + b_hh[512 + hid];
    float bi2 = b_ih[1024 + hid] + b_hh[1024 + hid];
    float bi3 = b_ih[1536 + hid] + b_hh[1536 + hid];
    int mhi = (lane >> 4) << 3;
#pragma unroll
    for (int r = 0; r < 8; ++r) {
        int m = m0 + mhi + r;
        float gi = sigm(acc0[r] + bi0);
        float gf = sigm(acc1[r] + bi1);
        float gg = tanhf(acc2[r] + bi2);
        float go = sigm(acc3[r] + bi3);
        long  ci = (long)m * H_ + hid;
        float cn = gf * c[ci] + gi * gg;
        c[ci] = cn;
        u16 hb = f2bf(go * tanhf(cn));
        h_next[ci] = hb;
        hs_out[(long)m * hs_stride + hid] = hb;
    }
}

// ---------------------------------------------------------------------------
// FC1: z = relu(hs1 @ fc1_w^T + b)  [43008,512] -> [43008,256], bf16 out.
__global__ __launch_bounds__(256) void fc1_kernel(const u16* __restrict__ A,
                                                  const u16* __restrict__ W,
                                                  const float* __restrict__ bias,
                                                  u16* __restrict__ Z) {
    int lane = threadIdx.x & 31;
    int wave = threadIdx.x >> 5;
    int tile = blockIdx.x * 8 + wave;
    int m0   = (tile >> 4) << 4;     // 2688 row tiles
    int n0   = (tile & 15) << 4;     // 16 col tiles
    v8f acc = {};
    for (int k0 = 0; k0 < H_; k0 += 32) {
        v16bf a = load_a_tile(A, H_, m0, k0, lane);
        v16bf b = load_b_tile(W, H_, n0, k0, lane);
        acc = WMMA_BF16(a, b, acc);
    }
    int n  = n0 + (lane & 15);
    float bi = bias[n];
    int mhi = (lane >> 4) << 3;
#pragma unroll
    for (int r = 0; r < 8; ++r) {
        int m = m0 + mhi + r;
        float v = acc[r] + bi;
        Z[(long)m * (H_ / 2) + n] = f2bf(v > 0.0f ? v : 0.0f);
    }
}

// FC2: out = z @ fc2_w^T + b  (N=1) — plain dot per row.
__global__ void fc2_kernel(const u16* __restrict__ Z, const float* __restrict__ w,
                           const float* __restrict__ b, float* __restrict__ out) {
    int row = blockIdx.x * 256 + threadIdx.x;
    const u16* zr = Z + (long)row * (H_ / 2);
    float s = b[0];
#pragma unroll 4
    for (int k = 0; k < H_ / 2; ++k) s += bf2f(zr[k]) * w[k];
    out[row] = s;
}

// ---------------------------------------------------------------------------
extern "C" void kernel_launch(void* const* d_in, const int* in_sizes, int n_in,
                              void* d_out, int out_size, void* d_ws, size_t ws_size,
                              hipStream_t stream) {
    (void)in_sizes; (void)n_in; (void)out_size; (void)ws_size;
    const int*   x        = (const int*)  d_in[0];
    const float* emb_day  = (const float*)d_in[1];
    const float* emb_mon  = (const float*)d_in[2];
    const float* emb_hour = (const float*)d_in[3];
    const float* emb_min  = (const float*)d_in[4];
    const float* W_ih0    = (const float*)d_in[5];
    const float* W_hh0    = (const float*)d_in[6];
    const float* b_ih0    = (const float*)d_in[7];
    const float* b_hh0    = (const float*)d_in[8];
    const float* W_ih1    = (const float*)d_in[9];
    const float* W_hh1    = (const float*)d_in[10];
    const float* b_ih1    = (const float*)d_in[11];
    const float* b_hh1    = (const float*)d_in[12];
    const float* fc1_w    = (const float*)d_in[13];
    const float* fc1_b    = (const float*)d_in[14];
    const float* fc2_w    = (const float*)d_in[15];
    const float* fc2_b    = (const float*)d_in[16];
    float* out = (float*)d_out;

    char* ws = (char*)d_ws;
    size_t off = 0;
    auto alloc = [&](size_t bytes) -> char* {
        char* p = ws + off;
        off += (bytes + 255) & ~(size_t)255;
        return p;
    };
    u16*   e     = (u16*)  alloc((size_t)B_ * T_ * I_ * 2);        // embeddings bf16
    u16*   hs0   = (u16*)  alloc((size_t)B_ * T_ * H_ * 2);        // layer0 outputs bf16
    u16*   hs1   = (u16*)  alloc((size_t)B_ * T_ * H_ * 2);        // layer1 outputs bf16
    u16*   z     = (u16*)  alloc((size_t)B_ * T_ * (H_ / 2) * 2);  // fc1 out bf16
    u16*   wih0  = (u16*)  alloc((size_t)4 * H_ * I_ * 2);
    u16*   whh0  = (u16*)  alloc((size_t)4 * H_ * H_ * 2);
    u16*   wih1  = (u16*)  alloc((size_t)4 * H_ * H_ * 2);
    u16*   whh1  = (u16*)  alloc((size_t)4 * H_ * H_ * 2);
    u16*   fw1   = (u16*)  alloc((size_t)(H_ / 2) * H_ * 2);
    u16*   hping = (u16*)  alloc((size_t)B_ * H_ * 2);
    u16*   hpong = (u16*)  alloc((size_t)B_ * H_ * 2);
    float* c     = (float*)alloc((size_t)B_ * H_ * 4);

    auto cvt = [&](const float* in, u16* o, int n) {
        cvt_bf16_kernel<<<(n + 255) / 256, 256, 0, stream>>>(in, o, n);
    };
    cvt(W_ih0, wih0, 4 * H_ * I_);
    cvt(W_hh0, whh0, 4 * H_ * H_);
    cvt(W_ih1, wih1, 4 * H_ * H_);
    cvt(W_hh1, whh1, 4 * H_ * H_);
    cvt(fc1_w, fw1, (H_ / 2) * H_);

    embed_kernel<<<B_ * T_, 256, 0, stream>>>(x, emb_day, emb_mon, emb_hour, emb_min, e);

    // ---- layer 0 scan ----
    u16 *hp = hping, *hn = hpong;
    zero_kernel<<<(B_ * H_ / 2 + 255) / 256, 256, 0, stream>>>((unsigned*)hp, B_ * H_ / 2);
    zero_kernel<<<(B_ * H_ + 255) / 256, 256, 0, stream>>>((unsigned*)c, B_ * H_);
    for (int t = 0; t < T_; ++t) {
        lstm_step_kernel<<<64, 256, 0, stream>>>(
            e + (size_t)t * I_, T_ * I_, I_, hp, wih0, whh0, b_ih0, b_hh0,
            c, hn, hs0 + (size_t)t * H_, T_ * H_);
        u16* tmp = hp; hp = hn; hn = tmp;
    }

    // ---- layer 1 scan ----
    zero_kernel<<<(B_ * H_ / 2 + 255) / 256, 256, 0, stream>>>((unsigned*)hp, B_ * H_ / 2);
    zero_kernel<<<(B_ * H_ + 255) / 256, 256, 0, stream>>>((unsigned*)c, B_ * H_);
    for (int t = 0; t < T_; ++t) {
        lstm_step_kernel<<<64, 256, 0, stream>>>(
            hs0 + (size_t)t * H_, T_ * H_, H_, hp, wih1, whh1, b_ih1, b_hh1,
            c, hn, hs1 + (size_t)t * H_, T_ * H_);
        u16* tmp = hp; hp = hn; hn = tmp;
    }

    // ---- FC head ----
    fc1_kernel<<<(B_ * T_ / 16) * 16 / 8, 256, 0, stream>>>(hs1, fw1, fc1_b, z);
    fc2_kernel<<<B_ * T_ / 256, 256, 0, stream>>>(z, fc2_w, fc2_b, out);
}
